// LSTMModule_22497038696966
// MI455X (gfx1250) — matrix operand order repaired and checked by reading, not compiled, not run on hardware
//
#include <hip/hip_runtime.h>

#define NWG   32
#define BDIM  128

constexpr int Bsz = 64;
constexpr int Tsz = 1024;
constexpr int Hsz = 512;
constexpr int Lsz = 2;
constexpr int G4H = 4 * Hsz;              // 2048 gate rows per layer
constexpr int WROW = 520;                 // padded LDS row stride (bf16 elements)
constexpr int LDS_BYTES = 2 * 64 * WROW * 2;  // 2 matrices * 64 rows * 520 * 2B = 133120

typedef __bf16 bf16;
typedef __bf16 v16bf __attribute__((ext_vector_type(16)));
typedef __bf16 v8bf  __attribute__((ext_vector_type(8)));
typedef __bf16 v4bf  __attribute__((ext_vector_type(4)));
typedef float  v8f   __attribute__((ext_vector_type(8)));
typedef float  v4f   __attribute__((ext_vector_type(4)));

__device__ __forceinline__ v16bf cat8(v8bf lo, v8bf hi) {
  v16bf r;
#pragma unroll
  for (int i = 0; i < 8; ++i) { r[i] = lo[i]; r[8 + i] = hi[i]; }
  return r;
}

// A-matrix tile (16x32 bf16) from a bf16 row in global memory.
// Lane l<16: row=l, K {k0..k0+7} U {k0+16..k0+23}; lanes>=16 shifted by 8 (k0 includes it).
__device__ __forceinline__ v16bf load_a_bf16(const bf16* rowp, int k0) {
  return cat8(*(const v8bf*)(rowp + k0), *(const v8bf*)(rowp + k0 + 16));
}

// Same A tile but sourced from fp32 (input x / h0), converted in-register to bf16.
__device__ __forceinline__ v16bf load_a_f32(const float* rowp, int k0) {
  v4f x0 = *(const v4f*)(rowp + k0);
  v4f x1 = *(const v4f*)(rowp + k0 + 4);
  v4f x2 = *(const v4f*)(rowp + k0 + 16);
  v4f x3 = *(const v4f*)(rowp + k0 + 20);
  v16bf a;
#pragma unroll
  for (int i = 0; i < 4; ++i) {
    a[i]      = (bf16)x0[i];
    a[4 + i]  = (bf16)x1[i];
    a[8 + i]  = (bf16)x2[i];
    a[12 + i] = (bf16)x3[i];
  }
  return a;
}

// B-matrix tile (32x16 bf16) from LDS-resident weight slice: K {k0..k0+7} U {k0+16..k0+23} per lane.
__device__ __forceinline__ v16bf ldsB(const bf16* p) {
  return cat8(*(const v8bf*)p, *(const v8bf*)(p + 16));
}

__device__ __forceinline__ float fast_sigmoid(float x) {
  return __builtin_amdgcn_rcpf(1.0f + __expf(-x));
}
__device__ __forceinline__ float fast_tanh(float x) {
  return 1.0f - 2.0f * __builtin_amdgcn_rcpf(__expf(2.0f * x) + 1.0f);
}

// Monotonic-counter grid barrier (persistent kernel, all NWG blocks resident).
__device__ __forceinline__ void grid_barrier(unsigned* cnt, unsigned target) {
  __syncthreads();
  if (threadIdx.x == 0) {
    __hip_atomic_fetch_add(cnt, 1u, __ATOMIC_ACQ_REL, __HIP_MEMORY_SCOPE_AGENT);
    while (__hip_atomic_load(cnt, __ATOMIC_ACQUIRE, __HIP_MEMORY_SCOPE_AGENT) < target) {
      __builtin_amdgcn_s_sleep(1);
    }
  }
  __syncthreads();
}

// 4 WMMAs (i,f,g,o gate tiles) against one A tile; B tiles from LDS weight slice.
__device__ __forceinline__ void mma4(v16bf a, const bf16* wb, int boff,
                                     v8f& acc0, v8f& acc1, v8f& acc2, v8f& acc3) {
  v16bf b0 = ldsB(wb + boff);
  v16bf b1 = ldsB(wb + boff + 16 * WROW);
  v16bf b2 = ldsB(wb + boff + 32 * WROW);
  v16bf b3 = ldsB(wb + boff + 48 * WROW);
  acc0 = __builtin_amdgcn_wmma_f32_16x16x32_bf16(false, a, false, b0, (short)0, acc0, false, false);
  acc1 = __builtin_amdgcn_wmma_f32_16x16x32_bf16(false, a, false, b1, (short)0, acc1, false, false);
  acc2 = __builtin_amdgcn_wmma_f32_16x16x32_bf16(false, a, false, b2, (short)0, acc2, false, false);
  acc3 = __builtin_amdgcn_wmma_f32_16x16x32_bf16(false, a, false, b3, (short)0, acc3, false, false);
}

__global__ void init_cnt(void* ws) {
  if (threadIdx.x == 0) *(unsigned*)ws = 0u;
}

__global__ void __launch_bounds__(BDIM, 1)
lstm_persistent(const float* __restrict__ x,
                const float* __restrict__ h0,
                const float* __restrict__ c0,
                const float* __restrict__ w_ih,
                const float* __restrict__ w_hh,
                const float* __restrict__ b_ih,
                const float* __restrict__ b_hh,
                float* __restrict__ out,
                void* ws) {
  extern __shared__ bf16 sW[];  // [2 matrices][64 rows][WROW] ; rows = 4 gates x 16 cols
  unsigned* cnt = (unsigned*)ws;
  bf16* hbuf = (bf16*)((char*)ws + 256);  // ping-pong [2][B*H] bf16

  const int wg   = blockIdx.x;       // owns hidden cols [wg*16, wg*16+16)
  const int tid  = threadIdx.x;
  const int wave = tid >> 5;         // owns batch rows [wave*16, wave*16+16)
  const int lane = tid & 31;
  const int n    = lane & 15;        // N col within tile / A row within tile
  const int kh   = lane >> 4;        // K-half select
  const int mbase = wave << 4;
  const int gcol  = (wg << 4) + n;   // global hidden column
  const int arow  = mbase + n;       // batch row for A-matrix loads
  const size_t OUT_H = (size_t)Bsz * Tsz * Hsz;
  const size_t OUT_C = OUT_H + (size_t)Lsz * Bsz * Hsz;

  unsigned bar = 0;
  float prevh[8];

  for (int l = 0; l < Lsz; ++l) {
    // ---- stage this WG's bf16 weight slices into LDS ----
    __syncthreads();
    {
      const float* wi = w_ih + (size_t)l * G4H * Hsz;
      const float* wh = w_hh + (size_t)l * G4H * Hsz;
      for (int c4 = tid; c4 < 16384; c4 += BDIM) {
        int e = c4 << 2;                  // element in unpadded [2][64][512]
        int m = e >> 15;
        int rowIdx = (e >> 9) & 63;       // gate*16 + r
        int k = e & 511;
        int g = rowIdx >> 4, r = rowIdx & 15;
        const float* src = (m ? wh : wi) + (size_t)(g * Hsz + (wg << 4) + r) * Hsz + k;
        v4f v = *(const v4f*)src;
        v4bf pk;
        pk[0] = (bf16)v[0]; pk[1] = (bf16)v[1]; pk[2] = (bf16)v[2]; pk[3] = (bf16)v[3];
        *(v4bf*)(sW + (size_t)(m * 64 + rowIdx) * WROW + k) = pk;
      }
    }
    __syncthreads();

    const bf16* sWih = sW;
    const bf16* sWhh = sW + 64 * WROW;

    // per-lane gate biases (uniform over batch rows)
    float bias[4];
#pragma unroll
    for (int g = 0; g < 4; ++g)
      bias[g] = b_ih[(size_t)l * G4H + g * Hsz + gcol] + b_hh[(size_t)l * G4H + g * Hsz + gcol];

    // persistent cell state in registers (C-tile layout: vgpr v -> row mbase+8*kh+v)
    v8f creg;
#pragma unroll
    for (int v = 0; v < 8; ++v)
      creg[v] = c0[(size_t)l * Bsz * Hsz + (size_t)(mbase + (kh << 3) + v) * Hsz + gcol];

    const float* xbase = (l == 0) ? x : out;  // [B,T,H] fp32

    for (int t = 0; t < Tsz; ++t) {
      // layer 1: delayed write of previous step's h into d_out (slab t-1, disjoint from slab t reads)
      if (l == 1 && t > 0) {
#pragma unroll
        for (int v = 0; v < 8; ++v)
          out[((size_t)(mbase + (kh << 3) + v) * Tsz + (t - 1)) * Hsz + gcol] = prevh[v];
      }

      v8f z = {0.f, 0.f, 0.f, 0.f, 0.f, 0.f, 0.f, 0.f};
      v8f acc0 = z, acc1 = z, acc2 = z, acc3 = z;

      // ---- input projection: gates += x_t[64,512] x W_ih_slice ----
      const float* xrow = xbase + ((size_t)arow * Tsz + t) * Hsz;
      if (t + 1 < Tsz) __builtin_prefetch(xrow + Hsz, 0, 0);
#pragma unroll 4
      for (int ks = 0; ks < 16; ++ks) {
        v16bf a = load_a_f32(xrow, (ks << 5) + (kh << 3));
        mma4(a, sWih, n * WROW + (ks << 5) + (kh << 3), acc0, acc1, acc2, acc3);
      }

      // ---- recurrence: gates += h_{t-1}[64,512] x W_hh_slice ----
      if (t == 0) {
        const float* hrow = h0 + (size_t)l * Bsz * Hsz + (size_t)arow * Hsz;
#pragma unroll 4
        for (int ks = 0; ks < 16; ++ks) {
          v16bf a = load_a_f32(hrow, (ks << 5) + (kh << 3));
          mma4(a, sWhh, n * WROW + (ks << 5) + (kh << 3), acc0, acc1, acc2, acc3);
        }
      } else {
        const bf16* hrow = hbuf + (size_t)((t - 1) & 1) * Bsz * Hsz + (size_t)arow * Hsz;
#pragma unroll 4
        for (int ks = 0; ks < 16; ++ks) {
          v16bf a = load_a_bf16(hrow, (ks << 5) + (kh << 3));
          mma4(a, sWhh, n * WROW + (ks << 5) + (kh << 3), acc0, acc1, acc2, acc3);
        }
      }

      // ---- elementwise LSTM cell, entirely in registers ----
      bf16* hw = hbuf + (size_t)(t & 1) * Bsz * Hsz;
#pragma unroll
      for (int v = 0; v < 8; ++v) {
        int brow = mbase + (kh << 3) + v;
        float ig = fast_sigmoid(acc0[v] + bias[0]);
        float fg = fast_sigmoid(acc1[v] + bias[1]);
        float gg = fast_tanh(acc2[v] + bias[2]);
        float og = fast_sigmoid(acc3[v] + bias[3]);
        float cc = fg * creg[v] + ig * gg;
        creg[v] = cc;
        float hh = og * fast_tanh(cc);
        hw[(size_t)brow * Hsz + gcol] = (bf16)hh;          // broadcast h for next step
        if (l == 0) out[((size_t)brow * Tsz + t) * Hsz + gcol] = hh;  // layer-1 input
        else        prevh[v] = hh;                          // delayed d_out store
        if (t == Tsz - 1) {
          out[OUT_H + (size_t)l * Bsz * Hsz + (size_t)brow * Hsz + gcol] = hh;
          out[OUT_C + (size_t)l * Bsz * Hsz + (size_t)brow * Hsz + gcol] = cc;
        }
      }

      __threadfence();
      ++bar;
      grid_barrier(cnt, bar * NWG);
    }

    if (l == 1) {  // flush last step's delayed slab
#pragma unroll
      for (int v = 0; v < 8; ++v)
        out[((size_t)(mbase + (kh << 3) + v) * Tsz + (Tsz - 1)) * Hsz + gcol] = prevh[v];
    }
  }
}

extern "C" void kernel_launch(void* const* d_in, const int* in_sizes, int n_in,
                              void* d_out, int out_size, void* d_ws, size_t ws_size,
                              hipStream_t stream) {
  (void)in_sizes; (void)n_in; (void)out_size; (void)ws_size;
  const float* x    = (const float*)d_in[0];
  const float* h0   = (const float*)d_in[1];
  const float* c0   = (const float*)d_in[2];
  const float* w_ih = (const float*)d_in[3];
  const float* w_hh = (const float*)d_in[4];
  const float* b_ih = (const float*)d_in[5];
  const float* b_hh = (const float*)d_in[6];
  float* out = (float*)d_out;

  init_cnt<<<1, 32, 0, stream>>>(d_ws);
  lstm_persistent<<<dim3(NWG), dim3(BDIM), LDS_BYTES, stream>>>(
      x, h0, c0, w_ih, w_hh, b_ih, b_hh, out, d_ws);
}
